// Topdown3DGCNHead_11742440588155
// MI455X (gfx1250) — compile-verified, matrix operand order checked by reading, not run on hardware
//
#include <hip/hip_runtime.h>

#define N_    32
#define K_    17
#define CIN_  2048
#define HW_   4096
#define W_    64
#define H_    64
#define PE_   64
#define CEMB_ 2112   // PE_ + CIN_
#define HID_  128
#define ROWS_ (N_*K_)  // 544 = 34*16

typedef __attribute__((ext_vector_type(16))) _Float16 v16h;
typedef __attribute__((ext_vector_type(8)))  _Float16 v8h;
typedef __attribute__((ext_vector_type(8)))  float    v8f;

// ---------------- argmax over heatmap [ROWS_, HW_] -> loc ----------------
__global__ void argmax_kernel(const float* __restrict__ hm, int* __restrict__ loc) {
    __shared__ float sval[256];
    __shared__ int   sidx[256];
    const int row = blockIdx.x;
    const float* p = hm + (size_t)row * HW_;
    const int t = threadIdx.x;
    float best = -1.0e30f; int bidx = 0;
    for (int i = t; i < HW_; i += 256) {
        float v = p[i];
        if (v > best) { best = v; bidx = i; }   // strided scan keeps lowest index on ties
    }
    sval[t] = best; sidx[t] = bidx;
    __syncthreads();
    for (int s = 128; s > 0; s >>= 1) {
        if (t < s) {
            float v2 = sval[t+s]; int i2 = sidx[t+s];
            if (v2 > sval[t] || (v2 == sval[t] && i2 < sidx[t])) { sval[t] = v2; sidx[t] = i2; }
        }
        __syncthreads();
    }
    if (t == 0) loc[row] = sidx[0];
}

// ------- weight prep: W[P, Kdim, Ncols] f32 -> WT[P, Ncols, Kdim] f16 (transposed) -------
__global__ void wprep_kernel(const float* __restrict__ Wsrc, _Float16* __restrict__ WT,
                             int Kdim, int Ncols, int P) {
    const size_t total = (size_t)P * Kdim * Ncols;
    for (size_t idx = (size_t)blockIdx.x * blockDim.x + threadIdx.x; idx < total;
         idx += (size_t)gridDim.x * blockDim.x) {
        const int    k = (int)(idx % Kdim);
        const size_t r = idx / Kdim;
        const int    n = (int)(r % Ncols);
        const int    p = (int)(r / Ncols);
        WT[idx] = (_Float16)Wsrc[((size_t)p * Kdim + k) * Ncols + n];
    }
}

// ------- embedding: posemb MLP + feature gather -> hin16 [ROWS_, CEMB_] (f16) -------
__global__ void embed_kernel(const float* __restrict__ x, const float* __restrict__ bbox,
                             const float* __restrict__ pos_w, const float* __restrict__ pos_b,
                             const int* __restrict__ loc, _Float16* __restrict__ hin16) {
    const int row = blockIdx.x;
    const int n = row / K_;
    const int l = loc[row];
    const float kx = (float)(l % W_);
    const float ky = (float)(l / W_);
    float in6[6];
    in6[0] = kx / (float)W_ * 2.0f - 1.0f;
    in6[1] = ky / (float)H_ * 2.0f - 1.0f;
    in6[2] = bbox[n*4+0] / 1920.0f * 2.0f - 1.0f;
    in6[3] = bbox[n*4+1] / 1080.0f * 2.0f - 1.0f;
    in6[4] = bbox[n*4+2] / 1920.0f;
    in6[5] = bbox[n*4+3] / 1080.0f;
    _Float16* out = hin16 + (size_t)row * CEMB_;
    const float* xr = x + (size_t)n * CIN_ * HW_ + l;
    for (int idx = threadIdx.x; idx < CEMB_; idx += blockDim.x) {
        if (idx < PE_) {
            float acc = pos_b[idx];
            #pragma unroll
            for (int c = 0; c < 6; ++c) acc += in6[c] * pos_w[c*PE_ + idx];
            out[idx] = (_Float16)acc;
        } else {
            out[idx] = (_Float16)xr[(size_t)(idx - PE_) * HW_];  // scattered gather
        }
    }
}

// ------------- WMMA GEMM: C = A16[M,Kdim] * WT16[path][Ncols,Kdim]^T -------------
// A and B both f16, both read as contiguous 16B chunks; one wave per 16x64 strip.
// Out-of-range columns are address-clamped only: their garbage stays in unstored D columns.
__global__ void gemm_wmma(const _Float16* __restrict__ A16, int lda,
                          const _Float16* __restrict__ WT,
                          float* __restrict__ C0, float* __restrict__ C1,
                          int ldc, int M, int Ncols, int Kdim) {
    const int m0 = blockIdx.x * 16;
    const int n0 = blockIdx.y * 64;
    const int path = blockIdx.z;
    const _Float16* Wp = WT + (size_t)path * Ncols * Kdim;
    float* C = (path == 0) ? C0 : C1;
    const int lane = threadIdx.x & 31;
    const int mr = lane & 15;
    const int hi = lane >> 4;
    // fragment element e maps to k = k0 + (e/8)*16 + hi*8 + e%8
    const _Float16* arow = A16 + (size_t)(m0 + mr) * lda + hi * 8;
    const _Float16* brow[4];
    #pragma unroll
    for (int t = 0; t < 4; ++t) {
        const int ncol = n0 + t*16 + mr;
        const int nc = (ncol < Ncols) ? ncol : (Ncols - 1);   // clamp address only
        brow[t] = Wp + (size_t)nc * Kdim + hi * 8;
    }
    v8f acc[4] = {{}, {}, {}, {}};
    for (int k0 = 0; k0 < Kdim; k0 += 32) {
        v16h a;
        {
            const v8h* ap = (const v8h*)(arow + k0);   // 16B aligned
            v8h lo = ap[0], hi2 = ap[2];
            #pragma unroll
            for (int p = 0; p < 8; ++p) { a[p] = lo[p]; a[8+p] = hi2[p]; }
        }
        __builtin_prefetch(arow + k0 + 32, 0, 0);       // global_prefetch_b8
        #pragma unroll
        for (int t = 0; t < 4; ++t) {
            v16h b;
            const v8h* bp = (const v8h*)(brow[t] + k0);
            v8h lo = bp[0], hi2 = bp[2];
            #pragma unroll
            for (int p = 0; p < 8; ++p) { b[p] = lo[p]; b[8+p] = hi2[p]; }
            acc[t] = __builtin_amdgcn_wmma_f32_16x16x32_f16(false, a, false, b,
                                                            (short)0, acc[t], false, false);
        }
    }
    #pragma unroll
    for (int t = 0; t < 4; ++t) {
        const int ncol = n0 + t*16 + mr;
        #pragma unroll
        for (int v = 0; v < 8; ++v) {
            int m = m0 + v + 8*hi;      // C/D layout: VGPR v -> M = v + 8*(lane>=16)
            if (ncol < Ncols && m < M) C[(size_t)m * ldc + ncol] = acc[t][v];
        }
    }
}

// ------------- masked row-softmax of edge logits E[17,17] -> adj -------------
__global__ void adj_kernel(const float* __restrict__ E, float* __restrict__ adj) {
    const int i = threadIdx.x;
    if (i >= K_) return;
    const int ea[16] = {0,1,2,0,4,5,0,7,8,9,8,11,12,8,14,15};
    const int eb[16] = {1,2,3,4,5,6,7,8,9,10,11,12,13,14,15,16};
    bool m[K_];
    #pragma unroll
    for (int j = 0; j < K_; ++j) m[j] = (j == i);
    #pragma unroll
    for (int e = 0; e < 16; ++e) {
        if (ea[e] == i) m[eb[e]] = true;
        if (eb[e] == i) m[ea[e]] = true;
    }
    float mx = -1.0e30f;
    for (int j = 0; j < K_; ++j) if (m[j]) mx = fmaxf(mx, E[i*K_+j]);
    float ex[K_]; float s = 0.0f;
    for (int j = 0; j < K_; ++j) { ex[j] = m[j] ? expf(E[i*K_+j] - mx) : 0.0f; s += ex[j]; }
    const float inv = 1.0f / s;
    for (int j = 0; j < K_; ++j) adj[i*K_+j] = ex[j] * inv;
}

// --------- adjacency mix + bias (+BN+ReLU) (+residual); writes f32 + optional f16 mirror ---------
__global__ void mix_bn_kernel(const float* __restrict__ h0, const float* __restrict__ h1,
                              const float* __restrict__ adj, const float* __restrict__ bias,
                              const float* __restrict__ gamma, const float* __restrict__ beta,
                              const float* __restrict__ resid, float* __restrict__ out,
                              _Float16* __restrict__ out16, int D, int applyBN) {
    const int row = blockIdx.x;
    const int b = row / K_, i = row % K_;
    const int d = threadIdx.x;
    if (d >= D) return;
    float acc = adj[i*K_+i] * h0[((size_t)b*K_ + i)*D + d];
    for (int j = 0; j < K_; ++j) {
        if (j == i) continue;
        acc += adj[i*K_+j] * h1[((size_t)b*K_ + j)*D + d];
    }
    acc += bias[d];
    if (applyBN) {
        const float s = rsqrtf(1.0f + 1e-5f);   // BN eval: mean=0, var=1
        acc = fmaxf(acc * gamma[d] * s + beta[d], 0.0f);
    }
    if (resid) acc += resid[(size_t)row * D + d];
    out[(size_t)row * D + d] = acc;
    if (out16) out16[(size_t)row * D + d] = (_Float16)acc;
}

extern "C" void kernel_launch(void* const* d_in, const int* in_sizes, int n_in,
                              void* d_out, int out_size, void* d_ws, size_t ws_size,
                              hipStream_t stream) {
    (void)in_sizes; (void)n_in; (void)out_size; (void)ws_size;
    const float* x     = (const float*)d_in[0];
    const float* heat  = (const float*)d_in[1];
    const float* bbox  = (const float*)d_in[2];
    const float* pos_w = (const float*)d_in[3];
    const float* pos_b = (const float*)d_in[4];
    const float* Wi    = (const float*)d_in[5];
    const float* bi    = (const float*)d_in[6];
    const float* Ei    = (const float*)d_in[7];
    const float* gi    = (const float*)d_in[8];
    const float* bti   = (const float*)d_in[9];
    const float* Wh    = (const float*)d_in[10];
    const float* bh    = (const float*)d_in[11];
    const float* Eh    = (const float*)d_in[12];
    const float* gh    = (const float*)d_in[13];
    const float* bth   = (const float*)d_in[14];
    const float* Wo    = (const float*)d_in[15];
    const float* bo    = (const float*)d_in[16];
    const float* Eo    = (const float*)d_in[17];
    float* out = (float*)d_out;

    // workspace carve-up (~6 MB)
    char* ws = (char*)d_ws;
    auto alignup = [](size_t v){ return (v + 255) & ~(size_t)255; };
    size_t off = 0;
    int*      loc   = (int*)(ws + off);      off = alignup(off + (size_t)ROWS_ * sizeof(int));
    _Float16* hin16 = (_Float16*)(ws + off); off = alignup(off + (size_t)ROWS_ * CEMB_ * sizeof(_Float16));
    float*    h0    = (float*)(ws + off);    off = alignup(off + (size_t)ROWS_ * HID_ * sizeof(float));
    float*    h1    = (float*)(ws + off);    off = alignup(off + (size_t)ROWS_ * HID_ * sizeof(float));
    float*    HA    = (float*)(ws + off);    off = alignup(off + (size_t)ROWS_ * HID_ * sizeof(float));
    float*    HB    = (float*)(ws + off);    off = alignup(off + (size_t)ROWS_ * HID_ * sizeof(float));
    float*    R     = (float*)(ws + off);    off = alignup(off + (size_t)ROWS_ * HID_ * sizeof(float));
    _Float16* HA16  = (_Float16*)(ws + off); off = alignup(off + (size_t)ROWS_ * HID_ * sizeof(_Float16));
    _Float16* HB16  = (_Float16*)(ws + off); off = alignup(off + (size_t)ROWS_ * HID_ * sizeof(_Float16));
    _Float16* R16   = (_Float16*)(ws + off); off = alignup(off + (size_t)ROWS_ * HID_ * sizeof(_Float16));
    float*    adjb  = (float*)(ws + off);    off = alignup(off + (size_t)K_ * K_ * sizeof(float));
    _Float16* WiT   = (_Float16*)(ws + off); off = alignup(off + (size_t)2 * HID_ * CEMB_ * sizeof(_Float16));
    _Float16* WhT   = (_Float16*)(ws + off); off = alignup(off + (size_t)16 * HID_ * HID_ * sizeof(_Float16));
    _Float16* WoT   = (_Float16*)(ws + off); off = alignup(off + (size_t)2 * 3 * HID_ * sizeof(_Float16));

    // weight transpose + f16 convert (graph-captured; deterministic each call)
    wprep_kernel<<<1056, 256, 0, stream>>>(Wi, WiT, CEMB_, HID_, 2);     // 540,672 elems
    wprep_kernel<<<512,  256, 0, stream>>>(Wh, WhT, HID_,  HID_, 16);    // 262,144 elems
    wprep_kernel<<<3,    256, 0, stream>>>(Wo, WoT, HID_,  3,    2);     // 768 elems

    // 1) keypoint argmax, 2) embeddings (f16)
    argmax_kernel<<<ROWS_, 256, 0, stream>>>(heat, loc);
    embed_kernel<<<ROWS_, 256, 0, stream>>>(x, bbox, pos_w, pos_b, loc, hin16);

    const dim3 gHid(ROWS_/16, HID_/64, 2);   // 34 x 2 x 2 waves

    // input gconv + BN + ReLU
    gemm_wmma<<<gHid, 32, 0, stream>>>(hin16, CEMB_, WiT, h0, h1, HID_, ROWS_, HID_, CEMB_);
    adj_kernel<<<1, 32, 0, stream>>>(Ei, adjb);
    mix_bn_kernel<<<ROWS_, HID_, 0, stream>>>(h0, h1, adjb, bi, gi, bti, nullptr,
                                              HA, HA16, HID_, 1);

    // 4 residual blocks of 2 gconvs each
    float* Hcur = HA; float* Hnxt = HB;
    _Float16* H16cur = HA16; _Float16* H16nxt = HB16;
    for (int l = 0; l < 4; ++l) {
        const int i0 = 2*l, i1 = 2*l + 1;
        gemm_wmma<<<gHid, 32, 0, stream>>>(H16cur, HID_, WhT + (size_t)i0*2*HID_*HID_,
                                           h0, h1, HID_, ROWS_, HID_, HID_);
        adj_kernel<<<1, 32, 0, stream>>>(Eh + (size_t)i0*K_*K_, adjb);
        mix_bn_kernel<<<ROWS_, HID_, 0, stream>>>(h0, h1, adjb, bh + (size_t)i0*HID_,
                                                  gh + (size_t)i0*HID_, bth + (size_t)i0*HID_,
                                                  nullptr, R, R16, HID_, 1);
        gemm_wmma<<<gHid, 32, 0, stream>>>(R16, HID_, WhT + (size_t)i1*2*HID_*HID_,
                                           h0, h1, HID_, ROWS_, HID_, HID_);
        adj_kernel<<<1, 32, 0, stream>>>(Eh + (size_t)i1*K_*K_, adjb);
        mix_bn_kernel<<<ROWS_, HID_, 0, stream>>>(h0, h1, adjb, bh + (size_t)i1*HID_,
                                                  gh + (size_t)i1*HID_, bth + (size_t)i1*HID_,
                                                  Hcur, Hnxt, H16nxt, HID_, 1);
        float* tf = Hcur; Hcur = Hnxt; Hnxt = tf;
        _Float16* th = H16cur; H16cur = H16nxt; H16nxt = th;
    }

    // output gconv (Ncols = 3, one padded 64-col strip), no BN/ReLU, write d_out
    gemm_wmma<<<dim3(ROWS_/16, 1, 2), 32, 0, stream>>>(H16cur, HID_, WoT, h0, h1, 3,
                                                       ROWS_, 3, HID_);
    adj_kernel<<<1, 32, 0, stream>>>(Eo, adjb);
    mix_bn_kernel<<<ROWS_, 32, 0, stream>>>(h0, h1, adjb, bo, nullptr, nullptr,
                                            nullptr, out, nullptr, 3, 0);
}
